// ASTGCNBlock_80796924772963
// MI455X (gfx1250) — matrix operand order repaired and checked by reading, not compiled, not run on hardware
//
#include <hip/hip_runtime.h>

// ---------------- constants ----------------
#define Bn 8
#define Nn 2048
#define Tt 12
#define FIN 2
#define FCH 64
#define FTM 64

typedef __attribute__((ext_vector_type(16))) __bf16 v16bf;
typedef __attribute__((ext_vector_type(8)))  float  v8f;

__device__ __forceinline__ unsigned short f2bf(float f) {
    union { float f; unsigned int u; } c; c.f = f;
    unsigned int u = c.u;
    unsigned int r = (u + 0x7FFFu + ((u >> 16) & 1u)) >> 16;
    return (unsigned short)r;
}

// ---------------- K1: temporal attention -> E (B,12,12) ----------------
__global__ void k_temporal_E(const float* __restrict__ x, const float* __restrict__ U1,
                             const float* __restrict__ U2, const float* __restrict__ U3,
                             const float* __restrict__ be, const float* __restrict__ Ve,
                             float* __restrict__ E) {
    int b = blockIdx.x, tid = threadIdx.x;
    // acc[0..11]=tmp f0 over t, [12..23]=tmp f1 over t, [24..35]=Q f0 over u, [36..47]=Q f1 over u
    float acc[48];
    #pragma unroll
    for (int i = 0; i < 48; i++) acc[i] = 0.f;
    const float* xb = x + (size_t)b * Nn * FIN * Tt;
    float u30 = U3[0], u31 = U3[1];
    for (int n = tid; n < Nn; n += 256) {
        const float* xv = xb + (size_t)n * 24;
        float u1 = U1[n], u20 = U2[n], u21 = U2[Nn + n];
        #pragma unroll
        for (int t = 0; t < 12; t++) {
            float x0 = xv[t], x1 = xv[12 + t];
            acc[t]      += u1 * x0;
            acc[12 + t] += u1 * x1;
            float r = u30 * x0 + u31 * x1;
            acc[24 + t] += u20 * r;
            acc[36 + t] += u21 * r;
        }
    }
    __shared__ float red[256][48];
    #pragma unroll
    for (int i = 0; i < 48; i++) red[tid][i] = acc[i];
    __syncthreads();
    __shared__ float stats[48];
    if (tid < 48) {
        float s = 0.f;
        for (int r = 0; r < 256; r++) s += red[r][tid];
        stats[tid] = s;
    }
    __syncthreads();
    __shared__ float sg[12][12];
    if (tid < 144) {
        int u = tid / 12, v = tid % 12;
        // prod_t[u][v] = sum_f tmp[u][f]*Q[f][v]
        float p = stats[u] * stats[24 + v] + stats[12 + u] * stats[36 + v];
        p += be[u * 12 + v];
        sg[u][v] = 1.f / (1.f + __expf(-p));
    }
    __syncthreads();
    __shared__ float Ep[12][12];
    if (tid < 144) {
        int t = tid / 12, v = tid % 12;
        float e = 0.f;
        #pragma unroll
        for (int u = 0; u < 12; u++) e += Ve[t * 12 + u] * sg[u][v];
        Ep[t][v] = e;
    }
    __syncthreads();
    __shared__ float mx[12], sm[12];
    if (tid < 12) {
        int v = tid;
        float m = -1e30f;
        for (int t = 0; t < 12; t++) m = fmaxf(m, Ep[t][v]);
        float s = 0.f;
        for (int t = 0; t < 12; t++) s += __expf(Ep[t][v] - m);
        mx[v] = m; sm[v] = 1.f / s;
    }
    __syncthreads();
    if (tid < 144) {
        int t = tid / 12, v = tid % 12;
        E[b * 144 + t * 12 + v] = __expf(Ep[t][v] - mx[v]) * sm[v];
    }
}

// ---------------- K2: x_TAt = x @ E ----------------
__global__ void k_apply_E(const float* __restrict__ x, const float* __restrict__ E,
                          float* __restrict__ xT) {
    __shared__ float Es[144];
    int gid = blockIdx.x * 256 + threadIdx.x;          // row index over (b,n,f): 32768 rows
    int b = gid >> 12;                                  // 4096 rows per batch
    if (threadIdx.x < 144) Es[threadIdx.x] = E[b * 144 + threadIdx.x];
    __syncthreads();
    const float* xr = x + (size_t)gid * 12;
    float xv[12];
    #pragma unroll
    for (int u = 0; u < 12; u++) xv[u] = xr[u];
    float* o = xT + (size_t)gid * 12;
    #pragma unroll
    for (int t = 0; t < 12; t++) {
        float s = 0.f;
        #pragma unroll
        for (int u = 0; u < 12; u++) s += xv[u] * Es[u * 12 + t];
        o[t] = s;
    }
}

// ---------------- K3: spatial attention features ----------------
__global__ void k_spatial_feats(const float* __restrict__ xT, const float* __restrict__ W1,
                                const float* __restrict__ W2, const float* __restrict__ W3,
                                float* __restrict__ lhs, float* __restrict__ rhsT) {
    int gid = blockIdx.x * 256 + threadIdx.x;   // b*2048+n
    const float* xv = xT + (size_t)gid * 24;
    float x0[12], x1[12];
    #pragma unroll
    for (int t = 0; t < 12; t++) { x0[t] = xv[t]; x1[t] = xv[12 + t]; }
    float a0 = 0.f, a1 = 0.f;
    #pragma unroll
    for (int t = 0; t < 12; t++) { a0 += x0[t] * W1[t]; a1 += x1[t] * W1[t]; }
    float w30 = W3[0], w31 = W3[1];
    #pragma unroll
    for (int t = 0; t < 12; t++) {
        lhs[(size_t)gid * 12 + t]  = a0 * W2[t] + a1 * W2[12 + t];
        rhsT[(size_t)gid * 12 + t] = w30 * x0[t] + w31 * x1[t];
    }
}

// ---------------- K0: f32 -> bf16 conversion ----------------
__global__ void k_cvt_bf16(const float* __restrict__ in, unsigned short* __restrict__ out, int n) {
    int gid = blockIdx.x * 256 + threadIdx.x;
    if (gid < n) out[gid] = f2bf(in[gid]);
}

// ---------------- K4: prod_s + bias + sigmoid -> bf16 ----------------
__global__ void k_prod_sig(const float* __restrict__ lhs, const float* __restrict__ rhsT,
                           const float* __restrict__ bs, unsigned short* __restrict__ sig) {
    int b = blockIdx.z;
    int i0 = blockIdx.y * 16;
    int j0 = blockIdx.x * 256;
    __shared__ float rs[256 * 12];
    __shared__ float ls[16 * 12];
    int tid = threadIdx.x;
    const float* rbase = rhsT + ((size_t)b * Nn + j0) * 12;
    for (int l = tid; l < 256 * 12; l += 256) rs[l] = rbase[l];
    const float* lbase = lhs + ((size_t)b * Nn + i0) * 12;
    if (tid < 192) ls[tid] = lbase[tid];
    __syncthreads();
    int j = j0 + tid;
    float rv[12];
    #pragma unroll
    for (int u = 0; u < 12; u++) rv[u] = rs[tid * 12 + u];
    for (int ii = 0; ii < 16; ii++) {
        float p = 0.f;
        #pragma unroll
        for (int u = 0; u < 12; u++) p += ls[ii * 12 + u] * rv[u];
        p += bs[(size_t)(i0 + ii) * Nn + j];
        float s = 1.f / (1.f + __expf(-p));
        sig[((size_t)b * Nn + i0 + ii) * Nn + j] = f2bf(s);
    }
}

// ---------------- K5: S_pre[b] = Vs(bf16) @ sig(bf16), f32 accumulate (WMMA) ----------------
// Block tile 128(M) x 64(N), K-step 32. 8 waves in a 4x2 grid; each wave owns a
// 2x2 grid of 16x16 accumulator tiles -> 4 WMMAs per K-step per wave.
__global__ void k_gemm_S(const unsigned short* __restrict__ A,   // Vs_bf [2048][2048]
                         const unsigned short* __restrict__ Bm,  // sig_bf [b][2048][2048]
                         float* __restrict__ C) {                // S_pre [b][2048][2048]
    int b  = blockIdx.z;
    int m0 = blockIdx.y * 128;
    int p0 = blockIdx.x * 64;
    const unsigned short* Bb = Bm + (size_t)b * Nn * Nn;
    float* Cb = C + (size_t)b * Nn * Nn;

    __shared__ unsigned int   As[128 * 16];  // 128 rows x 32 bf16 (16 dwords/row) = 8KB
    __shared__ unsigned short Bst[64 * 32];  // transposed: [p_local][k_local]    = 4KB

    int tid  = threadIdx.x;
    int wave = tid >> 5;
    int lane = tid & 31;
    int wr   = wave >> 1;         // wave row 0..3 -> M rows [wr*32, wr*32+32)
    int wc   = wave & 1;          // wave col 0..1 -> N cols [wc*32, wc*32+32)
    int l16  = lane & 15;
    int hi   = (lane >> 4) & 1;

    // staging source pointers (thread-invariant across k-loop except k offset)
    const unsigned int* Ag = (const unsigned int*)A;
    const unsigned int* Bg = (const unsigned int*)Bb;
    int ra  = tid >> 1;               // A row 0..127
    int ca  = (tid & 1) * 8;          // A dword col 0 or 8
    int rb  = tid >> 3;               // B k-row 0..31
    int cbw = (tid & 7) * 4;          // B dword col 0..28

    union Frag { v16bf v; unsigned int u[8]; };
    v8f c00 = {}; v8f c01 = {}; v8f c10 = {}; v8f c11 = {};

    for (int k0 = 0; k0 < Nn; k0 += 32) {
        const unsigned int* srcA = Ag + (size_t)(m0 + ra) * (Nn / 2) + (k0 >> 1) + ca;
        const unsigned int* srcB = Bg + (size_t)(k0 + rb) * (Nn / 2) + (p0 >> 1) + cbw;
        // stage A: rows m0..m0+127, cols k0..k0+31 (8 dwords per thread)
        {
            unsigned int* dst = &As[ra * 16 + ca];
            #pragma unroll
            for (int q = 0; q < 8; q++) dst[q] = srcA[q];
        }
        // stage B transposed: Bst[p_local][k_local]
        {
            #pragma unroll
            for (int q = 0; q < 4; q++) {
                unsigned int d = srcB[q];
                int p = (cbw + q) * 2;
                Bst[p * 32 + rb]       = (unsigned short)(d & 0xFFFFu);
                Bst[(p + 1) * 32 + rb] = (unsigned short)(d >> 16);
            }
        }
        // prefetch next K tile into cache hierarchy (global_prefetch_b8)
        if (k0 + 32 < Nn) {
            __builtin_prefetch((const void*)(srcA + 16), 0, 0);
            __builtin_prefetch((const void*)(srcB + 16 * (Nn / 2)), 0, 0);
        }
        __syncthreads();
        // A fragments: rows wr*32 + {0,16} + l16; VGPR j dword = (j<4 ? j : 4+j) + hi*4
        Frag a0, a1;
        {
            const unsigned int* Ar0 = &As[(wr * 32 + l16) * 16];
            const unsigned int* Ar1 = &As[(wr * 32 + 16 + l16) * 16];
            #pragma unroll
            for (int j = 0; j < 4; j++) {
                a0.u[j]     = Ar0[j + hi * 4];
                a0.u[4 + j] = Ar0[8 + j + hi * 4];
                a1.u[j]     = Ar1[j + hi * 4];
                a1.u[4 + j] = Ar1[8 + j + hi * 4];
            }
        }
        // B fragments: cols wc*32 + {0,16} + l16; VGPR j dword = (j<4 ? j : 4+j) + hi*4
        Frag b0f, b1f;
        {
            const unsigned int* Bp0 = (const unsigned int*)&Bst[(wc * 32 + l16) * 32];
            const unsigned int* Bp1 = (const unsigned int*)&Bst[(wc * 32 + 16 + l16) * 32];
            #pragma unroll
            for (int j = 0; j < 8; j++) {
                b0f.u[j] = Bp0[(j < 4 ? j : 4 + j) + hi * 4];
                b1f.u[j] = Bp1[(j < 4 ? j : 4 + j) + hi * 4];
            }
        }
        c00 = __builtin_amdgcn_wmma_f32_16x16x32_bf16(false, a0.v, false, b0f.v, (short)0, c00, false, false);
        c01 = __builtin_amdgcn_wmma_f32_16x16x32_bf16(false, a0.v, false, b1f.v, (short)0, c01, false, false);
        c10 = __builtin_amdgcn_wmma_f32_16x16x32_bf16(false, a1.v, false, b0f.v, (short)0, c10, false, false);
        c11 = __builtin_amdgcn_wmma_f32_16x16x32_bf16(false, a1.v, false, b1f.v, (short)0, c11, false, false);
        __syncthreads();
    }
    // store C: lane l, VGPR v -> M = v + hi*8, N = l16
    int mrow0 = m0 + wr * 32 + hi * 8;
    int mrow1 = mrow0 + 16;
    int pcol  = p0 + wc * 32 + l16;
    #pragma unroll
    for (int v = 0; v < 8; v++) {
        Cb[(size_t)(mrow0 + v) * Nn + pcol]      = c00[v];
        Cb[(size_t)(mrow0 + v) * Nn + pcol + 16] = c01[v];
        Cb[(size_t)(mrow1 + v) * Nn + pcol]      = c10[v];
        Cb[(size_t)(mrow1 + v) * Nn + pcol + 16] = c11[v];
    }
}

// ---------------- K6: softmax over m (axis=1) of S (b,m,p), in place ----------------
__global__ void k_softmax_col(float* __restrict__ S) {
    int b = blockIdx.y;
    int p = blockIdx.x * 256 + threadIdx.x;
    float* Sb = S + (size_t)b * Nn * Nn;
    float m = -1e30f;
    for (int i = 0; i < Nn; i++) m = fmaxf(m, Sb[(size_t)i * Nn + p]);
    float s = 0.f;
    for (int i = 0; i < Nn; i++) s += __expf(Sb[(size_t)i * Nn + p] - m);
    float inv = 1.f / s;
    for (int i = 0; i < Nn; i++) {
        size_t idx = (size_t)i * Nn + p;
        Sb[idx] = __expf(Sb[idx] - m) * inv;
    }
}

// ---------------- K7: cheb graph conv + Theta + relu -> gcn (B,N,64,12) ----------------
__global__ void k_cheb_gcn(const float* __restrict__ cheb, const float* __restrict__ S,
                           const float* __restrict__ x, const float* __restrict__ Theta,
                           float* __restrict__ gcn) {
    int b = blockIdx.y;
    int n = blockIdx.x * 256 + threadIdx.x;
    __shared__ float xs[128][24];
    __shared__ float th[384];   // Theta[k][f][o]
    for (int l = threadIdx.x; l < 384; l += 256) th[l] = Theta[l];
    float acc[3][24];
    #pragma unroll
    for (int k = 0; k < 3; k++)
        #pragma unroll
        for (int i = 0; i < 24; i++) acc[k][i] = 0.f;
    const float* Sb = S + (size_t)b * Nn * Nn;
    const float* xb = x + (size_t)b * Nn * 24;
    const size_t chebStride = (size_t)Nn * Nn;
    for (int m0 = 0; m0 < Nn; m0 += 128) {
        __syncthreads();
        for (int l = threadIdx.x; l < 128 * 24; l += 256)
            xs[l / 24][l % 24] = xb[(size_t)m0 * 24 + l];
        __syncthreads();
        for (int mm = 0; mm < 128; mm++) {
            size_t moff = (size_t)(m0 + mm) * Nn + n;
            float s  = Sb[moff];
            float a0 = cheb[moff] * s;
            float a1 = cheb[chebStride + moff] * s;
            float a2 = cheb[2 * chebStride + moff] * s;
            const float* xm = xs[mm];
            #pragma unroll
            for (int ft = 0; ft < 24; ft++) {
                float xv = xm[ft];
                acc[0][ft] += a0 * xv;
                acc[1][ft] += a1 * xv;
                acc[2][ft] += a2 * xv;
            }
        }
    }
    float* out = gcn + ((size_t)b * Nn + n) * 768;
    for (int o = 0; o < 64; o++) {
        float t00 = th[o],       t01 = th[64 + o];
        float t10 = th[128 + o], t11 = th[192 + o];
        float t20 = th[256 + o], t21 = th[320 + o];
        #pragma unroll
        for (int t = 0; t < 12; t++) {
            float g = acc[0][t] * t00 + acc[0][12 + t] * t01
                    + acc[1][t] * t10 + acc[1][12 + t] * t11
                    + acc[2][t] * t20 + acc[2][12 + t] * t21;
            out[o * 12 + t] = fmaxf(g, 0.f);
        }
    }
}

// ---------------- K8: tconv(1x3) + residual(1x1) + relu + layernorm -> out ----------------
__global__ void k_out(const float* __restrict__ gcn, const float* __restrict__ x,
                      const float* __restrict__ tw, const float* __restrict__ tb,
                      const float* __restrict__ rw, const float* __restrict__ rb,
                      const float* __restrict__ lng, const float* __restrict__ lnb,
                      float* __restrict__ out) {
    int n = blockIdx.x, b = blockIdx.y;
    __shared__ float w[64 * 64 * 3];   // 48 KB
    __shared__ float g[64 * 12];
    __shared__ float y[12][64];
    __shared__ float mu[12], rstd[12];
    int tid = threadIdx.x;
    for (int l = tid; l < 64 * 64 * 3; l += 256) w[l] = tw[l];
    const float* gsrc = gcn + ((size_t)b * Nn + n) * 768;
    for (int l = tid; l < 768; l += 256) g[l] = gsrc[l];
    __syncthreads();
    const float* xr = x + ((size_t)b * Nn + n) * 24;
    float x0[12], x1[12];
    #pragma unroll
    for (int t = 0; t < 12; t++) { x0[t] = xr[t]; x1[t] = xr[12 + t]; }
    for (int e = tid; e < 768; e += 256) {
        int ft = e / 12, t = e % 12;
        float acc = tb[ft];
        const float* wf = &w[ft * 192];
        for (int fc = 0; fc < 64; fc++) {
            const float* gf = &g[fc * 12];
            float left  = (t > 0)  ? gf[t - 1] : 0.f;
            float mid   = gf[t];
            float right = (t < 11) ? gf[t + 1] : 0.f;
            acc += wf[fc * 3 + 0] * left + wf[fc * 3 + 1] * mid + wf[fc * 3 + 2] * right;
        }
        float res = rb[ft] + rw[ft * 2 + 0] * x0[t] + rw[ft * 2 + 1] * x1[t];
        y[t][ft] = fmaxf(acc + res, 0.f);
    }
    __syncthreads();
    if (tid < 12) {
        float s = 0.f;
        for (int f = 0; f < 64; f++) s += y[tid][f];
        float m = s * (1.f / 64.f);
        float v = 0.f;
        for (int f = 0; f < 64; f++) { float d = y[tid][f] - m; v += d * d; }
        mu[tid] = m;
        rstd[tid] = rsqrtf(v * (1.f / 64.f) + 1e-5f);
    }
    __syncthreads();
    float* o = out + ((size_t)b * Nn + n) * 768;
    for (int e = tid; e < 768; e += 256) {
        int ft = e / 12, t = e % 12;
        o[ft * 12 + t] = (y[t][ft] - mu[t]) * rstd[t] * lng[ft] + lnb[ft];
    }
}

// ---------------- host launcher ----------------
extern "C" void kernel_launch(void* const* d_in, const int* in_sizes, int n_in,
                              void* d_out, int out_size, void* d_ws, size_t ws_size,
                              hipStream_t stream) {
    const float* x     = (const float*)d_in[0];
    const float* cheb  = (const float*)d_in[1];
    const float* Theta = (const float*)d_in[2];
    const float* U1    = (const float*)d_in[3];
    const float* U2    = (const float*)d_in[4];
    const float* U3    = (const float*)d_in[5];
    const float* be    = (const float*)d_in[6];
    const float* Ve    = (const float*)d_in[7];
    const float* W1    = (const float*)d_in[8];
    const float* W2    = (const float*)d_in[9];
    const float* W3    = (const float*)d_in[10];
    const float* bs    = (const float*)d_in[11];
    const float* Vs    = (const float*)d_in[12];
    const float* tw    = (const float*)d_in[13];
    const float* tb    = (const float*)d_in[14];
    const float* rw    = (const float*)d_in[15];
    const float* rb    = (const float*)d_in[16];
    const float* lng   = (const float*)d_in[17];
    const float* lnb   = (const float*)d_in[18];
    float* out = (float*)d_out;

    char* ws = (char*)d_ws;
    size_t off = 0;
    float* E = (float*)(ws + off);              off += 8192;                              // B*144 f32 (padded)
    float* xT = (float*)(ws + off);             off += (size_t)Bn * Nn * 24 * 4;          // x_TAt
    float* lhs = (float*)(ws + off);            off += (size_t)Bn * Nn * 12 * 4;          // lhs_s
    float* rhsT = (float*)(ws + off);           off += (size_t)Bn * Nn * 12 * 4;          // rhs_s transposed
    unsigned short* Vsb = (unsigned short*)(ws + off); off += (size_t)Nn * Nn * 2;        // Vs bf16
    unsigned short* sig = (unsigned short*)(ws + off); off += (size_t)Bn * Nn * Nn * 2;   // sigmoid bf16
    float* S = (float*)(ws + off);              off += (size_t)Bn * Nn * Nn * 4;          // S_pre / S
    float* gcn = (float*)(ws + off);            off += (size_t)Bn * Nn * 768 * 4;         // gcn

    k_temporal_E<<<Bn, 256, 0, stream>>>(x, U1, U2, U3, be, Ve, E);
    k_apply_E<<<(Bn * Nn * FIN) / 256, 256, 0, stream>>>(x, E, xT);
    k_spatial_feats<<<(Bn * Nn) / 256, 256, 0, stream>>>(xT, W1, W2, W3, lhs, rhsT);
    k_cvt_bf16<<<(Nn * Nn + 255) / 256, 256, 0, stream>>>(Vs, Vsb, Nn * Nn);
    k_prod_sig<<<dim3(Nn / 256, Nn / 16, Bn), 256, 0, stream>>>(lhs, rhsT, bs, sig);
    k_gemm_S<<<dim3(Nn / 64, Nn / 128, Bn), 256, 0, stream>>>(Vsb, sig, S);
    k_softmax_col<<<dim3(Nn / 256, Bn), 256, 0, stream>>>(S);
    k_cheb_gcn<<<dim3(Nn / 256, Bn), 256, 0, stream>>>(cheb, S, x, Theta, gcn);
    k_out<<<dim3(Nn, Bn), 256, 0, stream>>>(gcn, x, tw, tb, rw, rb, lng, lnb, out);
}
